// EncoderBlock_51831665328197
// MI455X (gfx1250) — matrix operand order repaired and checked by reading, not compiled, not run on hardware
//
#include <hip/hip_runtime.h>
#include <hip/hip_bf16.h>
#include <cstdint>

#define B_   8
#define S_   1024
#define D_   1024
#define H_   16
#define DK_  64
#define DFF_ 4096
#define EPS_ 1e-6f
#define NEG_ -1e9f

typedef __attribute__((ext_vector_type(16))) __bf16 v16bf;
typedef __attribute__((ext_vector_type(8)))  float  v8f;

union Frag {
  v16bf v;
  uint4 q[2];
  unsigned short s[16];
};

__device__ __forceinline__ unsigned short f2bf(float f) {
  unsigned int u = __float_as_uint(f);
  unsigned int r = u + 0x7FFFu + ((u >> 16) & 1u);   // round-to-nearest-even
  return (unsigned short)(r >> 16);
}
__device__ __forceinline__ unsigned int pack2(float a, float b) {
  return (unsigned int)f2bf(a) | ((unsigned int)f2bf(b) << 16);
}

// CDNA5 async global->LDS copy (ASYNCcnt-tracked); per-lane 16B chunk.
__device__ __forceinline__ void async_copy_b128(unsigned lds_byte_off, const void* gsrc) {
  asm volatile("global_load_async_to_lds_b128 %0, %1, off"
               :: "v"(lds_byte_off), "v"((unsigned long long)(uintptr_t)gsrc)
               : "memory");
}
__device__ __forceinline__ void wait_async0() {
  asm volatile("s_wait_asynccnt 0x0" ::: "memory");
}

// ---------------------------------------------------------------------------
// One-shot f32 -> bf16 conversion (activations & weights), 8 elems/thread.
// ---------------------------------------------------------------------------
__global__ __launch_bounds__(256)
void cvt_bf16(const float* __restrict__ in, unsigned short* __restrict__ out) {
  size_t i = ((size_t)blockIdx.x * 256 + threadIdx.x) * 8;
  float4 a = *(const float4*)(in + i);
  float4 b = *(const float4*)(in + i + 4);
  uint4 o;
  o.x = pack2(a.x, a.y); o.y = pack2(a.z, a.w);
  o.z = pack2(b.x, b.y); o.w = pack2(b.z, b.w);
  *(uint4*)(out + i) = o;
}

// ---------------------------------------------------------------------------
// GEMM: C[M,N] = A[M,K](bf16) * W[N,K](bf16)^T + bias, optional ReLU,
// f32 or bf16 output. 128x128 block, K-step 64, double-buffered async LDS.
// ---------------------------------------------------------------------------
#define TBM 128
#define TBN 128
#define TBK 64
#define LDA 72   // padded LDS row stride (halfwords): 144B, conflict-free, 16B-aligned

__global__ __launch_bounds__(256)
void gemm_bf16(const unsigned short* __restrict__ A, const unsigned short* __restrict__ W,
               const float* __restrict__ bias, float* __restrict__ Cf,
               unsigned short* __restrict__ Cb, int M, int N, int K, int relu)
{
  __shared__ __align__(16) unsigned short As[2][TBM * LDA];
  __shared__ __align__(16) unsigned short Bs[2][TBN * LDA];

  const int tid  = threadIdx.x;
  const int lane = tid & 31;
  const int wv   = tid >> 5;
  const int wm   = wv >> 1;        // 0..3 : 32-row strip
  const int wn   = wv & 1;         // 0..1 : 64-col strip
  const int bm   = blockIdx.y * TBM;
  const int bn   = blockIdx.x * TBN;
  const int mrow = lane & 15;
  const int half = lane >> 4;

  const unsigned asBase0 = (unsigned)(uintptr_t)&As[0][0];
  const unsigned asBase1 = (unsigned)(uintptr_t)&As[1][0];
  const unsigned bsBase0 = (unsigned)(uintptr_t)&Bs[0][0];
  const unsigned bsBase1 = (unsigned)(uintptr_t)&Bs[1][0];

  v8f acc[2][4];
  v8f zero = {0.f,0.f,0.f,0.f,0.f,0.f,0.f,0.f};
#pragma unroll
  for (int i = 0; i < 2; ++i)
#pragma unroll
    for (int j = 0; j < 4; ++j) acc[i][j] = zero;

  const int nk = K / TBK;
  // issue helper expanded inline (4 A-chunks + 4 B-chunks per thread per tile)
#define ISSUE_TILE(ABASE, BBASE, K0)                                          \
  {                                                                           \
    _Pragma("unroll")                                                         \
    for (int i_ = 0; i_ < 4; ++i_) {                                          \
      int id_  = i_ * 256 + tid;             /* 1024 16B-chunks per matrix */ \
      int row_ = id_ >> 3;                                                    \
      int c_   = id_ & 7;                                                     \
      unsigned loff_ = (unsigned)((row_ * LDA + c_ * 8) * 2);                 \
      async_copy_b128((ABASE) + loff_, A + (size_t)(bm + row_) * K + (K0) + c_ * 8); \
      async_copy_b128((BBASE) + loff_, W + (size_t)(bn + row_) * K + (K0) + c_ * 8); \
    }                                                                         \
  }

  ISSUE_TILE(asBase0, bsBase0, 0)
  int cur = 0;

  for (int kt = 0; kt < nk; ++kt) {
    wait_async0();
    __syncthreads();   // buffer 'cur' filled & visible; prev buffer fully consumed
    if (kt + 1 < nk) {
      if (cur == 0) ISSUE_TILE(asBase1, bsBase1, (kt + 1) * TBK)
      else          ISSUE_TILE(asBase0, bsBase0, (kt + 1) * TBK)
    }
    const unsigned short* Ab = As[cur];
    const unsigned short* Bb = Bs[cur];
#pragma unroll
    for (int kk = 0; kk < 2; ++kk) {
      Frag a[2];
#pragma unroll
      for (int mi = 0; mi < 2; ++mi) {
        const unsigned short* p = Ab + (wm * 32 + mi * 16 + mrow) * LDA + kk * 32 + half * 8;
        a[mi].q[0] = *(const uint4*)(p);
        a[mi].q[1] = *(const uint4*)(p + 16);
      }
#pragma unroll
      for (int ni = 0; ni < 4; ++ni) {
        Frag b;
        const unsigned short* p = Bb + (wn * 64 + ni * 16 + mrow) * LDA + kk * 32 + half * 8;
        b.q[0] = *(const uint4*)(p);
        b.q[1] = *(const uint4*)(p + 16);
#pragma unroll
        for (int mi = 0; mi < 2; ++mi)
          acc[mi][ni] = __builtin_amdgcn_wmma_f32_16x16x32_bf16(
              false, a[mi].v, false, b.v, (short)0, acc[mi][ni], false, false);
      }
    }
    cur ^= 1;
  }
#undef ISSUE_TILE

#pragma unroll
  for (int ni = 0; ni < 4; ++ni) {
    const int col = bn + wn * 64 + ni * 16 + mrow;
    const float bb = bias[col];
#pragma unroll
    for (int mi = 0; mi < 2; ++mi) {
      const int rbase = bm + wm * 32 + mi * 16 + half * 8;
#pragma unroll
      for (int v = 0; v < 8; ++v) {
        float val = acc[mi][ni][v] + bb;
        if (relu) val = fmaxf(val, 0.f);
        if (Cb) Cb[(size_t)(rbase + v) * N + col] = f2bf(val);
        else    Cf[(size_t)(rbase + v) * N + col] = val;
      }
    }
  }
}

// ---------------------------------------------------------------------------
// Flash-attention on bf16 Q/K/V: per block one (b,h) x 128 q rows, 8 waves.
// ---------------------------------------------------------------------------
#define KT  64
#define LDV 72

__global__ __launch_bounds__(256)
void attn_fa(const unsigned short* __restrict__ Qm, const unsigned short* __restrict__ Km,
             const unsigned short* __restrict__ Vm, const int* __restrict__ mask,
             unsigned short* __restrict__ O)
{
  __shared__ __align__(16) unsigned short Ks[KT  * LDV];
  __shared__ __align__(16) unsigned short Vt[DK_ * LDV];      // V transposed: [d][key]
  __shared__ __align__(16) unsigned short Ps[8 * 16 * LDV];   // per-wave P~ tiles

  const int tid  = threadIdx.x;
  const int lane = tid & 31;
  const int wv   = tid >> 5;
  const int nQB  = S_ / 128;
  const int qb   = blockIdx.x % nQB;
  const int h    = (blockIdx.x / nQB) % H_;
  const int b    = blockIdx.x / (nQB * H_);
  const int mrow = lane & 15;
  const int half = lane >> 4;
  const int q0   = qb * 128 + wv * 16;
  const unsigned ksBase = (unsigned)(uintptr_t)&Ks[0];

  // Q fragments: direct b128 loads of bf16 (scale folded in post-WMMA)
  Frag qf[2];
  {
    const size_t qbase = (size_t)(b * S_ + q0 + mrow) * D_ + h * DK_;
#pragma unroll
    for (int kk = 0; kk < 2; ++kk) {
      const unsigned short* p0 = Qm + qbase + kk * 32 + half * 8;
      qf[kk].q[0] = *(const uint4*)(p0);
      qf[kk].q[1] = *(const uint4*)(p0 + 16);
    }
  }

  v8f zero = {0.f,0.f,0.f,0.f,0.f,0.f,0.f,0.f};
  v8f o[4];
#pragma unroll
  for (int t = 0; t < 4; ++t) o[t] = zero;
  float mrun[8], lrun[8];
#pragma unroll
  for (int v = 0; v < 8; ++v) { mrun[v] = -1e30f; lrun[v] = 0.f; }

  for (int s0 = 0; s0 < S_; s0 += KT) {
    __syncthreads();  // previous tile fully consumed
    // K tile: async copy (512 x 16B chunks); V tile: transpose via u16 scatter
#pragma unroll
    for (int i = 0; i < 2; ++i) {
      int id  = i * 256 + tid;
      int row = id >> 3;
      int c   = id & 7;
      size_t g = (size_t)(b * S_ + s0 + row) * D_ + h * DK_ + c * 8;
      async_copy_b128(ksBase + (unsigned)((row * LDV + c * 8) * 2), Km + g);
      Frag vv;
      vv.q[0] = *(const uint4*)(Vm + g);
#pragma unroll
      for (int j = 0; j < 8; ++j)
        Vt[(c * 8 + j) * LDV + row] = vv.s[j];
    }
    wait_async0();
    __syncthreads();

    // S = (Q K^T) * 1/sqrt(DK), then mask
    v8f sacc[4];
#pragma unroll
    for (int t = 0; t < 4; ++t) {
      sacc[t] = zero;
#pragma unroll
      for (int kk = 0; kk < 2; ++kk) {
        Frag bk;
        const unsigned short* p = &Ks[(t * 16 + mrow) * LDV + kk * 32 + half * 8];
        bk.q[0] = *(const uint4*)(p);
        bk.q[1] = *(const uint4*)(p + 16);
        sacc[t] = __builtin_amdgcn_wmma_f32_16x16x32_bf16(
            false, qf[kk].v, false, bk.v, (short)0, sacc[t], false, false);
      }
      int mv = mask[b * S_ + s0 + t * 16 + mrow];
      if (mv == 0) {
#pragma unroll
        for (int v = 0; v < 8; ++v) sacc[t][v] = NEG_;
      } else {
#pragma unroll
        for (int v = 0; v < 8; ++v) sacc[t][v] *= 0.125f;
      }
    }

    // online softmax (rows: VGPR idx + lane half; cols: lanes 0..15)
    float tmax[8];
#pragma unroll
    for (int v = 0; v < 8; ++v) {
      float m0 = fmaxf(fmaxf(sacc[0][v], sacc[1][v]), fmaxf(sacc[2][v], sacc[3][v]));
#pragma unroll
      for (int off = 1; off < 16; off <<= 1)
        m0 = fmaxf(m0, __shfl_xor(m0, off, 16));
      tmax[v] = m0;
    }
    float al[8], ps[8];
#pragma unroll
    for (int v = 0; v < 8; ++v) {
      float mn = fmaxf(mrun[v], tmax[v]);
      al[v] = __expf(mrun[v] - mn);
      mrun[v] = mn;
      ps[v] = 0.f;
    }
#pragma unroll
    for (int t = 0; t < 4; ++t)
#pragma unroll
      for (int v = 0; v < 8; ++v) {
        float p = __expf(sacc[t][v] - mrun[v]);
        sacc[t][v] = p;
        ps[v] += p;
      }
#pragma unroll
    for (int v = 0; v < 8; ++v) {
#pragma unroll
      for (int off = 1; off < 16; off <<= 1)
        ps[v] += __shfl_xor(ps[v], off, 16);
      lrun[v] = lrun[v] * al[v] + ps[v];
    }
#pragma unroll
    for (int t = 0; t < 4; ++t)
#pragma unroll
      for (int v = 0; v < 8; ++v) o[t][v] *= al[v];

    // P~ -> per-wave LDS tile (C-layout -> A-fragment layout)
    unsigned short* pw = &Ps[wv * 16 * LDV];
#pragma unroll
    for (int t = 0; t < 4; ++t)
#pragma unroll
      for (int v = 0; v < 8; ++v)
        pw[(half * 8 + v) * LDV + t * 16 + mrow] = f2bf(sacc[t][v]);
    __syncthreads();

    // O += P~ * V
#pragma unroll
    for (int td = 0; td < 4; ++td) {
#pragma unroll
      for (int kk = 0; kk < 2; ++kk) {
        Frag pa;
        const unsigned short* pp = &Ps[wv * 16 * LDV + mrow * LDV + kk * 32 + half * 8];
        pa.q[0] = *(const uint4*)(pp);
        pa.q[1] = *(const uint4*)(pp + 16);
        Frag vb;
        const unsigned short* pv = &Vt[(td * 16 + mrow) * LDV + kk * 32 + half * 8];
        vb.q[0] = *(const uint4*)(pv);
        vb.q[1] = *(const uint4*)(pv + 16);
        o[td] = __builtin_amdgcn_wmma_f32_16x16x32_bf16(
            false, pa.v, false, vb.v, (short)0, o[td], false, false);
      }
    }
  }

  // normalize + store context as bf16
#pragma unroll
  for (int v = 0; v < 8; ++v) lrun[v] = 1.f / lrun[v];
#pragma unroll
  for (int td = 0; td < 4; ++td)
#pragma unroll
    for (int v = 0; v < 8; ++v)
      O[(size_t)(b * S_ + q0 + half * 8 + v) * D_ + h * DK_ + td * 16 + mrow] =
          f2bf(o[td][v] * lrun[v]);
}

// ---------------------------------------------------------------------------
// Fused residual + LayerNorm (ddof=1, eps added to sigma); optional bf16 copy.
// ---------------------------------------------------------------------------
__global__ __launch_bounds__(256)
void ln_residual(const float* __restrict__ A, const float* __restrict__ R,
                 const float* __restrict__ g, const float* __restrict__ be,
                 float* __restrict__ Out, unsigned short* __restrict__ OutB)
{
  __shared__ float sh[2][8];
  const int tid = threadIdx.x;
  const size_t base = (size_t)blockIdx.x * D_;
  float vals[4];
  float sum = 0.f, sq = 0.f;
#pragma unroll
  for (int i = 0; i < 4; ++i) {
    int c = tid + i * 256;
    float x = A[base + c] + R[base + c];
    vals[i] = x; sum += x; sq += x * x;
  }
#pragma unroll
  for (int off = 16; off >= 1; off >>= 1) {
    sum += __shfl_xor(sum, off, 32);
    sq  += __shfl_xor(sq,  off, 32);
  }
  if ((tid & 31) == 0) { sh[0][tid >> 5] = sum; sh[1][tid >> 5] = sq; }
  __syncthreads();
  float ts = 0.f, tq = 0.f;
#pragma unroll
  for (int i = 0; i < 8; ++i) { ts += sh[0][i]; tq += sh[1][i]; }
  const float mu  = ts / (float)D_;
  float var = (tq - (float)D_ * mu * mu) / (float)(D_ - 1);
  var = fmaxf(var, 0.f);
  const float denom = EPS_ + sqrtf(var);
  const float gg = g[0], bb = be[0];
#pragma unroll
  for (int i = 0; i < 4; ++i) {
    int c = tid + i * 256;
    float y = gg * (vals[i] - mu) / denom + bb;
    Out[base + c] = y;
    if (OutB) OutB[base + c] = f2bf(y);
  }
}

// ---------------------------------------------------------------------------
extern "C" void kernel_launch(void* const* d_in, const int* in_sizes, int n_in,
                              void* d_out, int out_size, void* d_ws, size_t ws_size,
                              hipStream_t stream)
{
  (void)in_sizes; (void)n_in; (void)out_size; (void)ws_size;
  const float* x    = (const float*)d_in[0];
  const int*   mask = (const int*)  d_in[1];
  const float* Wq = (const float*)d_in[2];  const float* bq = (const float*)d_in[3];
  const float* Wk = (const float*)d_in[4];  const float* bk = (const float*)d_in[5];
  const float* Wv = (const float*)d_in[6];  const float* bv = (const float*)d_in[7];
  const float* Wo = (const float*)d_in[8];  const float* bo = (const float*)d_in[9];
  const float* W1 = (const float*)d_in[10]; const float* b1 = (const float*)d_in[11];
  const float* W2 = (const float*)d_in[12]; const float* b2 = (const float*)d_in[13];
  const float* g1 = (const float*)d_in[14]; const float* be1= (const float*)d_in[15];
  const float* g2 = (const float*)d_in[16]; const float* be2= (const float*)d_in[17];
  float* out = (float*)d_out;

  char* W8 = (char*)d_ws;
  const size_t MB = 1024ull * 1024ull;
  unsigned short* qb   = (unsigned short*)(W8 + 0);
  unsigned short* kb   = (unsigned short*)(W8 + 16 * MB);
  unsigned short* vb   = (unsigned short*)(W8 + 32 * MB);
  unsigned short* ctxb = (unsigned short*)(W8 + 48 * MB);
  float*          x1   = (float*)         (W8 + 64 * MB);
  unsigned short* x1b  = (unsigned short*)(W8 + 96 * MB);
  float*          ff2  = (float*)         (W8 + 112 * MB);
  unsigned short* xb   = (unsigned short*)(W8 + 144 * MB);
  unsigned short* Wqb  = (unsigned short*)(W8 + 160 * MB);
  unsigned short* Wkb  = (unsigned short*)(W8 + 162 * MB);
  unsigned short* Wvb  = (unsigned short*)(W8 + 164 * MB);
  unsigned short* Wob  = (unsigned short*)(W8 + 166 * MB);
  unsigned short* W1b  = (unsigned short*)(W8 + 168 * MB);
  unsigned short* W2b  = (unsigned short*)(W8 + 176 * MB);
  float*          proj = (float*)         (W8 + 0);       // reuse qb+kb (dead)
  unsigned short* ff1b = (unsigned short*)(W8 + 0);       // reuse 0..64MB (dead)

  const int M = B_ * S_;
  dim3 blk(256);

  // one-shot bf16 conversions (grid = elems / 2048; all sizes divide evenly)
  cvt_bf16<<<dim3((B_*S_*D_) / 2048), blk, 0, stream>>>(x,  xb);
  cvt_bf16<<<dim3((D_*D_)    / 2048), blk, 0, stream>>>(Wq, Wqb);
  cvt_bf16<<<dim3((D_*D_)    / 2048), blk, 0, stream>>>(Wk, Wkb);
  cvt_bf16<<<dim3((D_*D_)    / 2048), blk, 0, stream>>>(Wv, Wvb);
  cvt_bf16<<<dim3((D_*D_)    / 2048), blk, 0, stream>>>(Wo, Wob);
  cvt_bf16<<<dim3((DFF_*D_)  / 2048), blk, 0, stream>>>(W1, W1b);
  cvt_bf16<<<dim3((D_*DFF_)  / 2048), blk, 0, stream>>>(W2, W2b);

  gemm_bf16<<<dim3(D_ / TBN, M / TBM), blk, 0, stream>>>(xb, Wqb, bq, nullptr, qb, M, D_, D_, 0);
  gemm_bf16<<<dim3(D_ / TBN, M / TBM), blk, 0, stream>>>(xb, Wkb, bk, nullptr, kb, M, D_, D_, 0);
  gemm_bf16<<<dim3(D_ / TBN, M / TBM), blk, 0, stream>>>(xb, Wvb, bv, nullptr, vb, M, D_, D_, 0);

  attn_fa<<<dim3(B_ * H_ * (S_ / 128)), blk, 0, stream>>>(qb, kb, vb, mask, ctxb);

  gemm_bf16<<<dim3(D_ / TBN, M / TBM), blk, 0, stream>>>(ctxb, Wob, bo, proj, nullptr, M, D_, D_, 0);
  ln_residual<<<dim3(M), blk, 0, stream>>>(x, proj, g1, be1, x1, x1b);

  gemm_bf16<<<dim3(DFF_ / TBN, M / TBM), blk, 0, stream>>>(x1b, W1b, b1, nullptr, ff1b, M, DFF_, D_, 1);
  gemm_bf16<<<dim3(D_ / TBN, M / TBM), blk, 0, stream>>>(ff1b, W2b, b2, ff2, nullptr, M, D_, DFF_, 0);
  ln_residual<<<dim3(M), blk, 0, stream>>>(x1, ff2, g2, be2, out, nullptr);
}
